// SelfAttentionHead_35158602285795
// MI455X (gfx1250) — matrix operand order repaired and checked by reading, not compiled, not run on hardware
//
#include <hip/hip_runtime.h>

typedef __attribute__((ext_vector_type(16))) _Float16 v16h;
typedef __attribute__((ext_vector_type(8)))  _Float16 v8h;
typedef __attribute__((ext_vector_type(4)))  _Float16 v4h;
typedef __attribute__((ext_vector_type(8)))  float    v8f;
typedef int v4i_t __attribute__((vector_size(16)));   // b128 payload type

#define DEV static __device__ __forceinline__

static constexpr int BB = 8;
static constexpr int SS = 2048;
static constexpr int DD = 1024;
static constexpr int MM = BB * SS;       // 16384
static constexpr int LDPAD = DD + 8;     // 1032 halves: 2064B row stride
                                         // == 4 (mod 64) DWORD banks -> conflict-free frags

// ---------------------------------------------------------------------------
// Async global -> LDS staging (CDNA5 ASYNCcnt path), with safe fallback.
// Builtin signature (from toolchain diagnostic): param 1 is v4i in AS1
// (global source), so we pass (AS1 v4i*, AS3 v4i*, imm, imm).
// LDS offset = low 32 bits of the generic pointer (aperture layout).
// ---------------------------------------------------------------------------
DEV void cp_g2l_b128(void* ldst, const void* gsrc) {
#if defined(__gfx1250__) && __has_builtin(__builtin_amdgcn_global_load_async_to_lds_b128)
  __builtin_amdgcn_global_load_async_to_lds_b128(
      (__attribute__((address_space(1))) v4i_t*)(uintptr_t)gsrc,
      (__attribute__((address_space(3))) v4i_t*)(unsigned)(uintptr_t)ldst,
      0, 0);
#else
  *(v8h*)ldst = *(const v8h*)gsrc;       // synchronous fallback
#endif
}

DEV void cp_g2l_join() {
#if defined(__gfx1250__) && __has_builtin(__builtin_amdgcn_global_load_async_to_lds_b128)
#if __has_builtin(__builtin_amdgcn_s_wait_asynccnt)
  __builtin_amdgcn_s_wait_asynccnt(0);
#else
  asm volatile("s_wait_asynccnt 0" ::: "memory");
#endif
#endif
}

// ---------------------------------------------------------------------------
// WMMA fragment loaders, matching the CDNA5 ISA 16-bit A/B VGPR layouts.
// A (16x32, MxK) from a row-major [*,ld] tile:
//   lane l: m = l&15, kb = 8*(l>>4); holds K = kb..kb+7 (v16h[0..7]) and
//   K = kb+16..kb+23 (v16h[8..15]).  Two contiguous 16B loads.
// B (32x16, KxN) from a row-major *transposed* [N,ld] tile (Bt[n][k]):
//   lane l: n = l&15, kb = 8*(l>>4); holds K = kb..kb+7 (v16h[0..7]) and
//   K = kb+16..kb+23 (v16h[8..15]), same as A.
// Work for both global and LDS tiles (addrspace inferred after inlining).
// ---------------------------------------------------------------------------
DEV v16h load_frag_a(const _Float16* __restrict__ tile, int ld) {
  const int lane = threadIdx.x & 31;
  const int m  = lane & 15;
  const int kb = (lane >> 4) << 3;
  const _Float16* p = tile + (size_t)m * ld + kb;
  v8h lo = *(const v8h*)(p);
  v8h hi = *(const v8h*)(p + 16);
  v16h r;
#pragma unroll
  for (int t = 0; t < 8; ++t) { r[t] = lo[t]; r[8 + t] = hi[t]; }
  return r;
}

DEV v16h load_frag_b(const _Float16* __restrict__ tileT, int ld) {
  const int lane = threadIdx.x & 31;
  const int n  = lane & 15;
  const int kb = (lane >> 4) << 3;
  const _Float16* p = tileT + (size_t)n * ld + kb;
  v8h lo = *(const v8h*)(p);
  v8h hi = *(const v8h*)(p + 16);
  v16h r;
#pragma unroll
  for (int t = 0; t < 8; ++t) { r[t] = lo[t]; r[8 + t] = hi[t]; }
  return r;
}

DEV v8f wmma_f16(v16h a, v16h b, v8f c) {
  return __builtin_amdgcn_wmma_f32_16x16x32_f16(
      /*neg_a=*/false, a, /*neg_b=*/false, b,
      /*c_mod=*/(short)0, c, /*reuse_a=*/false, /*reuse_b=*/false);
}

// ---------------------------------------------------------------------------
// Per-wave GEMM core: 32 rows x 64 cols, K-loop in steps of 32.
// A: row-major [M,K]. BT: row-major [N,K] (i.e. B transposed).
// ---------------------------------------------------------------------------
DEV void gemm_core_32x64(const _Float16* __restrict__ A,
                         const _Float16* __restrict__ BT,
                         int K, int row0, int col0, v8f acc[2][4]) {
  const _Float16* pa0 = A + (size_t)row0 * K;
  const _Float16* pa1 = A + (size_t)(row0 + 16) * K;
  const _Float16* pb0 = BT + (size_t)(col0 +  0) * K;
  const _Float16* pb1 = BT + (size_t)(col0 + 16) * K;
  const _Float16* pb2 = BT + (size_t)(col0 + 32) * K;
  const _Float16* pb3 = BT + (size_t)(col0 + 48) * K;
  for (int k0 = 0; k0 < K; k0 += 32) {
    v16h a0 = load_frag_a(pa0 + k0, K);
    v16h a1 = load_frag_a(pa1 + k0, K);
    v16h b0 = load_frag_b(pb0 + k0, K);
    v16h b1 = load_frag_b(pb1 + k0, K);
    v16h b2 = load_frag_b(pb2 + k0, K);
    v16h b3 = load_frag_b(pb3 + k0, K);
    __builtin_prefetch(pa0 + k0 + 128, 0, 1);   // -> global_prefetch_b8
    __builtin_prefetch(pb0 + k0 + 128, 0, 1);
    acc[0][0] = wmma_f16(a0, b0, acc[0][0]);
    acc[0][1] = wmma_f16(a0, b1, acc[0][1]);
    acc[0][2] = wmma_f16(a0, b2, acc[0][2]);
    acc[0][3] = wmma_f16(a0, b3, acc[0][3]);
    acc[1][0] = wmma_f16(a1, b0, acc[1][0]);
    acc[1][1] = wmma_f16(a1, b1, acc[1][1]);
    acc[1][2] = wmma_f16(a1, b2, acc[1][2]);
    acc[1][3] = wmma_f16(a1, b3, acc[1][3]);
  }
}

// ---------------------------------------------------------------------------
// Kernel: f32 -> f16 convert (vectorized x4)
// ---------------------------------------------------------------------------
__global__ __launch_bounds__(256)
void k_cvt_x(const float* __restrict__ x, _Float16* __restrict__ y, int nvec) {
  int i = blockIdx.x * blockDim.x + threadIdx.x;
  if (i < nvec) {
    float4 v = ((const float4*)x)[i];
    v4h h;
    h[0] = (_Float16)v.x; h[1] = (_Float16)v.y;
    h[2] = (_Float16)v.z; h[3] = (_Float16)v.w;
    ((v4h*)y)[i] = h;
  }
}

// f32 [K,N] -> f16 transposed [N,K]
__global__ __launch_bounds__(256)
void k_cvt_wT(const float* __restrict__ W, _Float16* __restrict__ WT,
              int Kdim, int Ndim) {
  int idx = blockIdx.x * blockDim.x + threadIdx.x;
  if (idx < Kdim * Ndim) {
    int k = idx / Ndim;
    int n = idx - k * Ndim;
    WT[(size_t)n * Kdim + k] = (_Float16)W[idx];
  }
}

// ---------------------------------------------------------------------------
// Kernel: GEMM with +bias epilogue, f16 output (QKV projections).
// grid = (M/128, N/128), block = 256 (8 waves, 4x2 wave grid)
// ---------------------------------------------------------------------------
__global__ __launch_bounds__(256)
void k_gemm_bias_f16(const _Float16* __restrict__ A, const _Float16* __restrict__ BT,
                     const float* __restrict__ bias, _Float16* __restrict__ out,
                     int M, int N, int K) {
  const int wave = threadIdx.x >> 5, lane = threadIdx.x & 31;
  const int row0 = blockIdx.x * 128 + (wave & 3) * 32;
  const int col0 = blockIdx.y * 128 + (wave >> 2) * 64;
  v8f z = {};
  v8f acc[2][4];
#pragma unroll
  for (int r = 0; r < 2; ++r)
#pragma unroll
    for (int c = 0; c < 4; ++c) acc[r][c] = z;

  gemm_core_32x64(A, BT, K, row0, col0, acc);

  const int n = lane & 15, mh = (lane >> 4) << 3;
#pragma unroll
  for (int r = 0; r < 2; ++r)
#pragma unroll
    for (int c = 0; c < 4; ++c) {
      const int col = col0 + c * 16 + n;
      const float bv = bias[col];
#pragma unroll
      for (int e = 0; e < 8; ++e) {
        const int row = row0 + r * 16 + mh + e;
        out[(size_t)row * N + col] = (_Float16)(acc[r][c][e] + bv);
      }
    }
}

// ---------------------------------------------------------------------------
// Kernel: output GEMM, epilogue out = leaky_relu(rowsum[row]*acc + bias[col])
// ---------------------------------------------------------------------------
__global__ __launch_bounds__(256)
void k_gemm_out(const _Float16* __restrict__ V, const _Float16* __restrict__ WlT,
                const float* __restrict__ bias, const float* __restrict__ rowsum,
                float* __restrict__ out, int M, int N, int K) {
  const int wave = threadIdx.x >> 5, lane = threadIdx.x & 31;
  const int row0 = blockIdx.x * 128 + (wave & 3) * 32;
  const int col0 = blockIdx.y * 128 + (wave >> 2) * 64;
  v8f z = {};
  v8f acc[2][4];
#pragma unroll
  for (int r = 0; r < 2; ++r)
#pragma unroll
    for (int c = 0; c < 4; ++c) acc[r][c] = z;

  gemm_core_32x64(V, WlT, K, row0, col0, acc);

  const int n = lane & 15, mh = (lane >> 4) << 3;
#pragma unroll
  for (int r = 0; r < 2; ++r)
#pragma unroll
    for (int c = 0; c < 4; ++c) {
      const int col = col0 + c * 16 + n;
      const float bv = bias[col];
#pragma unroll
      for (int e = 0; e < 8; ++e) {
        const int row = row0 + r * 16 + mh + e;
        float val = rowsum[row] * acc[r][c][e] + bv;
        out[(size_t)row * N + col] = (val >= 0.0f) ? val : 0.01f * val;
      }
    }
}

// ---------------------------------------------------------------------------
// Kernel: column sums of exp(qk/1024), softmax denominators (axis=i).
// Block owns (b, 16 columns j). The 16-row Q block (32KB) is staged once in
// LDS via async global->LDS, then reused by all 8 waves x 16 i-tiles.
// qk[i,j] = <k_i, q_j>  -> A = K rows (global stream), B^T = Q rows (LDS).
// ---------------------------------------------------------------------------
__global__ __launch_bounds__(256)
void k_colsum(const _Float16* __restrict__ Kf, const _Float16* __restrict__ Qf,
              float* __restrict__ colsum, int S, int D) {
  const int b  = blockIdx.y;
  const int j0 = blockIdx.x * 16;
  const _Float16* Kb = Kf + (size_t)b * S * D;
  const _Float16* Qblk = Qf + ((size_t)b * S + j0) * D;

  __shared__ __align__(16) _Float16 qs[16][LDPAD];
  // 16 rows x 1024 halves = 2048 x 16B chunks; 8 chunks per thread
#pragma unroll
  for (int kk = 0; kk < 8; ++kk) {
    const int c = threadIdx.x + kk * 256;
    const int row = c >> 7;
    const int col = (c & 127) << 3;
    cp_g2l_b128(&qs[row][col], Qblk + (size_t)row * D + col);
  }
  cp_g2l_join();
  __syncthreads();

  const int wave = threadIdx.x >> 5, lane = threadIdx.x & 31;
  const int n = lane & 15;
  const float invT = 1.0f / 1024.0f;

  float cp = 0.0f;
  for (int it = 0; it < 16; ++it) {
    const int i0 = wave * 256 + it * 16;
    const _Float16* pk = Kb + (size_t)i0 * D;
    v8f acc = {};
    for (int d = 0; d < D; d += 32) {
      v16h a  = load_frag_a(pk + d, D);           // global stream
      v16h bq = load_frag_b(&qs[0][d], LDPAD);    // LDS, ds_load_b128
      __builtin_prefetch(pk + d + 128, 0, 1);
      acc = wmma_f16(a, bq, acc);
    }
#pragma unroll
    for (int e = 0; e < 8; ++e) cp += __expf(acc[e] * invT);
  }

  __shared__ float cacc[16];
  if (threadIdx.x < 16) cacc[threadIdx.x] = 0.0f;
  __syncthreads();
  atomicAdd(&cacc[n], cp);           // ds_add_f32
  __syncthreads();
  if (threadIdx.x < 16)
    colsum[(size_t)b * S + j0 + threadIdx.x] = cacc[threadIdx.x];
}

// ---------------------------------------------------------------------------
// Kernel: rowsum[i] = sum_j mask_j * exp(qk_ij/1024) / colsum_j + S*EPS
// Block owns (b, 16 rows i). The 16-row K block is staged once in LDS
// (A operand); Q rows stream from global (each read once per wave).
// ---------------------------------------------------------------------------
__global__ __launch_bounds__(256)
void k_rowsum(const _Float16* __restrict__ Kf, const _Float16* __restrict__ Qf,
              const float* __restrict__ mask, const float* __restrict__ colsum,
              float* __restrict__ rowsum, int S, int D) {
  const int b  = blockIdx.y;
  const int i0 = blockIdx.x * 16;
  const _Float16* Qb = Qf + (size_t)b * S * D;
  const _Float16* Kblk = Kf + ((size_t)b * S + i0) * D;

  __shared__ __align__(16) _Float16 ks[16][LDPAD];
#pragma unroll
  for (int kk = 0; kk < 8; ++kk) {
    const int c = threadIdx.x + kk * 256;
    const int row = c >> 7;
    const int col = (c & 127) << 3;
    cp_g2l_b128(&ks[row][col], Kblk + (size_t)row * D + col);
  }
  cp_g2l_join();
  __syncthreads();

  const int wave = threadIdx.x >> 5, lane = threadIdx.x & 31;
  const int n = lane & 15, mh = (lane >> 4) << 3;
  const float invT = 1.0f / 1024.0f;

  float rp[8];
#pragma unroll
  for (int e = 0; e < 8; ++e) rp[e] = 0.0f;

  for (int jt = 0; jt < 16; ++jt) {
    const int j0 = wave * 256 + jt * 16;
    const _Float16* pq = Qb + (size_t)j0 * D;
    v8f acc = {};
    for (int d = 0; d < D; d += 32) {
      v16h a  = load_frag_a(&ks[0][d], LDPAD);    // LDS, ds_load_b128
      v16h bq = load_frag_b(pq + d, D);           // global stream
      __builtin_prefetch(pq + d + 128, 0, 1);
      acc = wmma_f16(a, bq, acc);
    }
    const int j = j0 + n;
    const float fac = mask[(size_t)b * S + j] / colsum[(size_t)b * S + j];
#pragma unroll
    for (int e = 0; e < 8; ++e) rp[e] += __expf(acc[e] * invT) * fac;
  }

  __shared__ float racc[16];
  if (threadIdx.x < 16) racc[threadIdx.x] = 0.0f;
  __syncthreads();
#pragma unroll
  for (int e = 0; e < 8; ++e) atomicAdd(&racc[mh + e], rp[e]);
  __syncthreads();
  if (threadIdx.x < 16)
    rowsum[(size_t)b * S + i0 + threadIdx.x] =
        racc[threadIdx.x] + (float)S * 1e-14f;
}

// ---------------------------------------------------------------------------
extern "C" void kernel_launch(void* const* d_in, const int* in_sizes, int n_in,
                              void* d_out, int out_size, void* d_ws, size_t ws_size,
                              hipStream_t stream) {
  (void)in_sizes; (void)n_in; (void)out_size; (void)ws_size;

  const float* xs   = (const float*)d_in[0];
  const float* mask = (const float*)d_in[1];
  const float* Wk   = (const float*)d_in[2];
  const float* bk   = (const float*)d_in[3];
  const float* Wq   = (const float*)d_in[4];
  const float* bq   = (const float*)d_in[5];
  const float* Wv   = (const float*)d_in[6];
  const float* bv   = (const float*)d_in[7];
  const float* Wl   = (const float*)d_in[8];
  const float* bl   = (const float*)d_in[9];
  float* out = (float*)d_out;

  const size_t XSZ = (size_t)MM * DD;   // 16,777,216 elements
  const size_t WSZ = (size_t)DD * DD;   // 1,048,576 elements

  char* ws = (char*)d_ws;
  size_t off = 0;
  _Float16* xh  = (_Float16*)(ws + off); off += XSZ * 2;
  _Float16* kh  = (_Float16*)(ws + off); off += XSZ * 2;
  _Float16* qh  = (_Float16*)(ws + off); off += XSZ * 2;
  _Float16* vh  = (_Float16*)(ws + off); off += XSZ * 2;
  _Float16* WkT = (_Float16*)(ws + off); off += WSZ * 2;
  _Float16* WqT = (_Float16*)(ws + off); off += WSZ * 2;
  _Float16* WvT = (_Float16*)(ws + off); off += WSZ * 2;
  _Float16* WlT = (_Float16*)(ws + off); off += WSZ * 2;
  float* colsum = (float*)(ws + off); off += (size_t)MM * 4;
  float* rowsum = (float*)(ws + off); off += (size_t)MM * 4;

  // 0) convert activations + transpose-convert weights to f16
  {
    int nvec = (int)(XSZ / 4);
    k_cvt_x<<<nvec / 256, 256, 0, stream>>>(xs, xh, nvec);
    int wb = (int)(WSZ / 256);
    k_cvt_wT<<<wb, 256, 0, stream>>>(Wk, WkT, DD, DD);
    k_cvt_wT<<<wb, 256, 0, stream>>>(Wq, WqT, DD, DD);
    k_cvt_wT<<<wb, 256, 0, stream>>>(Wv, WvT, DD, DD);
    k_cvt_wT<<<wb, 256, 0, stream>>>(Wl, WlT, DD, DD);
  }

  // 1) QKV projection GEMMs (WMMA f16 -> f32 acc -> f16 store)
  dim3 gg(MM / 128, DD / 128);   // (128, 8)
  k_gemm_bias_f16<<<gg, 256, 0, stream>>>(xh, WkT, bk, kh, MM, DD, DD);
  k_gemm_bias_f16<<<gg, 256, 0, stream>>>(xh, WqT, bq, qh, MM, DD, DD);
  k_gemm_bias_f16<<<gg, 256, 0, stream>>>(xh, WvT, bv, vh, MM, DD, DD);

  // 2) softmax denominators (column sums over i), flash-style QK recompute
  dim3 gs(SS / 16, BB);          // (128, 8)
  k_colsum<<<gs, 256, 0, stream>>>(kh, qh, colsum, SS, DD);

  // 3) masked row sums of the normalized attention
  k_rowsum<<<gs, 256, 0, stream>>>(kh, qh, mask, colsum, rowsum, SS, DD);

  // 4) out = leaky_relu(rowsum .* (v @ Wl) + bl)
  k_gemm_out<<<gg, 256, 0, stream>>>(vh, WlT, bl, rowsum, out, MM, DD, DD);
}
